// LSTMBaseline_74655121539440
// MI455X (gfx1250) — matrix operand order repaired and checked by reading, not compiled, not run on hardware
//
#include <hip/hip_runtime.h>
#include <hip/hip_bf16.h>

// ---------------------------------------------------------------------------
// Problem constants (from reference)
// ---------------------------------------------------------------------------
constexpr int Bb = 64;     // batch
constexpr int Cc = 22;     // input channels
constexpr int Tt = 2560;   // time steps
constexpr int U0 = 128;    // layer-0 units
constexpr int U1 = 64;     // layer-1 units

// Workspace layout (bytes)
constexpr size_t OFF_XT  = 0;                                   // f16 [B][T][32]
constexpr size_t OFF_H0  = OFF_XT  + (size_t)Bb*Tt*32*2;        // f16 [B][T][256]
constexpr size_t OFF_WH0 = OFF_H0  + (size_t)Bb*Tt*256*2;       // f16 [2][512][128]
constexpr size_t OFF_WX0 = OFF_WH0 + (size_t)2*512*128*2;       // f16 [2][512][32]
constexpr size_t OFF_WX1 = OFF_WX0 + (size_t)2*512*32*2;        // f16 [2][256][256]
constexpr size_t OFF_WH1 = OFF_WX1 + (size_t)2*256*256*2;       // f16 [2][256][64]
constexpr size_t OFF_H1  = OFF_WH1 + (size_t)2*256*64*2;        // f32 [2][64][64]

// ---------------------------------------------------------------------------
// WMMA fragment types & loaders (gfx1250 wave32 layouts, CDNA5 ISA)
// ---------------------------------------------------------------------------
typedef __attribute__((ext_vector_type(16))) _Float16 v16h;
typedef __attribute__((ext_vector_type(8)))  float    v8f;

union F16x16 { v16h v; uint4 q[2]; };

// A fragment: 16x32 f16 (MxK). Row-major source, row = &A[m][0] (per-lane m).
// Lane<16 holds K=[0..15], lane>=16 holds K=[16..31]:
// two contiguous 16B loads per lane.
__device__ __forceinline__ v16h load_a16(const _Float16* row, int k0, int lh) {
  F16x16 f;
  f.q[0] = *(const uint4*)(row + k0 + (lh << 4));
  f.q[1] = *(const uint4*)(row + k0 + 8 + (lh << 4));
  return f.v;
}

// B fragment: 32x16 f16 (KxN) from an N-major (transposed) weight matrix
// WT[n][k], row length ldk halves. Lane n<16: K=k0..k0+15; lane>=16: +16.
__device__ __forceinline__ v16h load_b16(const _Float16* WT, int ldk, int n0,
                                         int k0, int lane) {
  const int n  = n0 + (lane & 15);
  const int kb = k0 + ((lane >> 4) << 4);
  const _Float16* p = WT + (size_t)n * ldk + kb;
  F16x16 f;
  f.q[0] = *(const uint4*)(p);
  f.q[1] = *(const uint4*)(p + 8);
  return f.v;
}

__device__ __forceinline__ v8f wmma_f16(v16h a, v16h b, v8f c) {
  return __builtin_amdgcn_wmma_f32_16x16x32_f16(false, a, false, b,
                                                (short)0, c, false, false);
}

// ---------------------------------------------------------------------------
// Fast nonlinearities: CDNA5 has a hardware v_tanh_f32 (TRANS unit, co-executes
// with the XDL/WMMA pipe). sigmoid(x) = 0.5*tanh(x/2)+0.5 -> 1 TRANS + 1 FMA.
// Fallback avoids IEEE divide sequences via v_exp_f32 + v_rcp_f32.
// ---------------------------------------------------------------------------
__device__ __forceinline__ float fast_rcp(float x) {
#if __has_builtin(__builtin_amdgcn_rcpf)
  return __builtin_amdgcn_rcpf(x);
#else
  return 1.0f / x;
#endif
}

__device__ __forceinline__ float fast_tanh(float x) {
#if __has_builtin(__builtin_amdgcn_tanhf)
  return __builtin_amdgcn_tanhf(x);
#else
  float xx = fminf(fmaxf(x, -15.f), 15.f);
  float e  = __expf(2.0f * xx);           // v_exp_f32 (+scale)
  return 1.0f - 2.0f * fast_rcp(e + 1.0f);
#endif
}

__device__ __forceinline__ float sigm(float x) {
  return fmaf(0.5f, fast_tanh(0.5f * x), 0.5f);
}

__device__ __forceinline__ void copy_to_lds(_Float16* dst, const _Float16* src,
                                            int nHalves) {
  const uint4* s = (const uint4*)src;
  uint4*       d = (uint4*)dst;
  const int n = nHalves >> 3;
  for (int i = threadIdx.x; i < n; i += blockDim.x) d[i] = s[i];
}

// ---------------------------------------------------------------------------
// Prep kernels
// ---------------------------------------------------------------------------
// x [B][C][T] f32  ->  xT [B][T][32] f16 (K padded 22->32 with zeros)
__global__ void prep_x_kernel(const float* __restrict__ x,
                              _Float16* __restrict__ xT) {
  int idx = blockIdx.x * blockDim.x + threadIdx.x;
  if (idx >= Bb * Tt) return;
  const int b = idx / Tt, t = idx % Tt;
  const float* xp = x + (size_t)b * Cc * Tt + t;   // stride Tt per channel
  _Float16* dst = xT + (size_t)idx * 32;
  #pragma unroll
  for (int c = 0; c < 32; ++c)
    dst[c] = (c < Cc) ? (_Float16)xp[(size_t)c * Tt] : (_Float16)0.0f;
}

// src f32 [K][N] row-major  ->  dst f16 [N][Kpad] (N-major, K padded w/ zeros)
__global__ void transpose_cast_kernel(const float* __restrict__ src,
                                      _Float16* __restrict__ dst,
                                      int K, int N, int Kpad) {
  int idx = blockIdx.x * blockDim.x + threadIdx.x;
  if (idx >= N * Kpad) return;
  const int n = idx / Kpad, k = idx % Kpad;
  dst[idx] = (k < K) ? (_Float16)src[(size_t)k * N + n] : (_Float16)0.0f;
}

// ---------------------------------------------------------------------------
// Layer-0 bidirectional LSTM: grid.x = direction (0 fwd, 1 bwd), 512 threads
//   z[64x512] = x_t[64x32]*WxT + h[64x128]*WhT + b ; gates ; h -> LDS + h0_seq
// 16 waves, each owns 2 (m,u) tile-groups x 4 gate tiles (all in registers).
// ---------------------------------------------------------------------------
__global__ __launch_bounds__(512) void lstm0_kernel(
    const _Float16* __restrict__ xT,     // [B][T][32]
    const _Float16* __restrict__ wh0T,   // [2][512][128]
    const _Float16* __restrict__ wx0T,   // [2][512][32]
    const float*    __restrict__ bias_f, // [512]
    const float*    __restrict__ bias_b, // [512]
    _Float16*       __restrict__ h0_seq) // [B][T][256]
{
  __shared__ _Float16 sWh[512 * 128];    // 128 KB, resident across all steps
  __shared__ _Float16 sH[64 * 128];      // 16 KB  current h (f16)

  const int dir = blockIdx.x;
  const _Float16* whT  = wh0T + (size_t)dir * 512 * 128;
  const _Float16* wxT  = wx0T + (size_t)dir * 512 * 32;
  const float*    bias = dir ? bias_b : bias_f;

  copy_to_lds(sWh, whT, 512 * 128);
  {
    uint4 z; z.x = z.y = z.z = z.w = 0u;
    for (int i = threadIdx.x; i < (64 * 128) / 8; i += blockDim.x)
      ((uint4*)sH)[i] = z;
  }
  __syncthreads();

  const int lane = threadIdx.x & 31;
  const int wv   = threadIdx.x >> 5;     // 0..15
  const int ut   = wv & 7;               // u tile (0..7), N-slice within gate
  const int ln   = lane & 15;
  const int lh   = lane >> 4;
  const int m0g[2] = { (wv >> 3) * 16, (wv >> 3) * 16 + 32 };

  float bvs[4];
  #pragma unroll
  for (int g = 0; g < 4; ++g) bvs[g] = bias[g * 128 + ut * 16 + ln];

  float c[2][8];
  #pragma unroll
  for (int grp = 0; grp < 2; ++grp)
    #pragma unroll
    for (int r = 0; r < 8; ++r) c[grp][r] = 0.0f;

  for (int t = 0; t < Tt; ++t) {
    const int td = dir ? (Tt - 1 - t) : t;

    v8f acc[2][4];
    #pragma unroll
    for (int grp = 0; grp < 2; ++grp)
      #pragma unroll
      for (int g = 0; g < 4; ++g)
        #pragma unroll
        for (int r = 0; r < 8; ++r) acc[grp][g][r] = bvs[g];

    // ---- input projection: K = 0..31 (one k-step), Wx from global (L2-hot)
    {
      v16h a[2];
      #pragma unroll
      for (int grp = 0; grp < 2; ++grp) {
        const int m = m0g[grp] + ln;
        a[grp] = load_a16(xT + ((size_t)m * Tt + td) * 32, 0, lh);
      }
      #pragma unroll
      for (int g = 0; g < 4; ++g) {
        v16h bf = load_b16(wxT, 32, g * 128 + ut * 16, 0, lane);
        #pragma unroll
        for (int grp = 0; grp < 2; ++grp)
          acc[grp][g] = wmma_f16(a[grp], bf, acc[grp][g]);
      }
    }
    // ---- recurrent: K = 128, 4 k-steps, Wh from LDS
    #pragma unroll
    for (int kk = 0; kk < 4; ++kk) {
      const int k0 = kk * 32;
      v16h a[2];
      #pragma unroll
      for (int grp = 0; grp < 2; ++grp)
        a[grp] = load_a16(sH + (size_t)(m0g[grp] + ln) * 128, k0, lh);
      #pragma unroll
      for (int g = 0; g < 4; ++g) {
        v16h bf = load_b16(sWh, 128, g * 128 + ut * 16, k0, lane);
        #pragma unroll
        for (int grp = 0; grp < 2; ++grp)
          acc[grp][g] = wmma_f16(a[grp], bf, acc[grp][g]);
      }
    }
    __syncthreads();   // all reads of sH for step t complete

    // ---- gates (element-aligned across the 4 accumulators), update c/h
    #pragma unroll
    for (int grp = 0; grp < 2; ++grp) {
      const int mb   = m0g[grp] + 8 * lh;
      const int ncol = ut * 16 + ln;
      #pragma unroll
      for (int r = 0; r < 8; ++r) {
        const float ig = sigm(acc[grp][0][r]);
        const float fg = sigm(acc[grp][1][r]);
        const float gg = fast_tanh(acc[grp][2][r]);
        const float og = sigm(acc[grp][3][r]);
        const float cc = fg * c[grp][r] + ig * gg;
        c[grp][r] = cc;
        const float hv = og * fast_tanh(cc);
        const _Float16 hh = (_Float16)hv;
        sH[(size_t)(mb + r) * 128 + ncol] = hh;
        h0_seq[((size_t)(mb + r) * Tt + td) * 256 + dir * 128 + ncol] = hh;
      }
    }
    __syncthreads();   // new h visible before step t+1
  }
}

// ---------------------------------------------------------------------------
// Layer-1 bidirectional LSTM (last state only): grid.x = direction
//   z[64x256] = h0_t[64x256]*Wx1T + h[64x64]*Wh1T + b
// 16 waves, each owns 1 (m,u) tile-group x 4 gate tiles.
// ---------------------------------------------------------------------------
__global__ __launch_bounds__(512) void lstm1_kernel(
    const _Float16* __restrict__ h0_seq, // [B][T][256]
    const _Float16* __restrict__ wx1T,   // [2][256][256]
    const _Float16* __restrict__ wh1T,   // [2][256][64]
    const float*    __restrict__ bias_f, // [256]
    const float*    __restrict__ bias_b, // [256]
    float*          __restrict__ h1_out) // [2][64][64]
{
  __shared__ _Float16 sWx[256 * 256];    // 128 KB input weights, resident
  __shared__ _Float16 sH[64 * 64];       // 8 KB  current h

  const int dir = blockIdx.x;
  const _Float16* whT  = wh1T + (size_t)dir * 256 * 64;
  const float*    bias = dir ? bias_b : bias_f;

  copy_to_lds(sWx, wx1T + (size_t)dir * 256 * 256, 256 * 256);
  {
    uint4 z; z.x = z.y = z.z = z.w = 0u;
    for (int i = threadIdx.x; i < (64 * 64) / 8; i += blockDim.x)
      ((uint4*)sH)[i] = z;
  }
  __syncthreads();

  const int lane = threadIdx.x & 31;
  const int wv   = threadIdx.x >> 5;     // 0..15
  const int m0   = (wv >> 2) * 16;       // 4 m tiles
  const int ut   = wv & 3;               // 4 u tiles
  const int ln   = lane & 15;
  const int lh   = lane >> 4;

  float bvs[4];
  #pragma unroll
  for (int g = 0; g < 4; ++g) bvs[g] = bias[g * 64 + ut * 16 + ln];

  float c[8];
  #pragma unroll
  for (int r = 0; r < 8; ++r) c[r] = 0.0f;

  for (int t = 0; t < Tt; ++t) {
    const int td = dir ? (Tt - 1 - t) : t;

    v8f acc[4];
    #pragma unroll
    for (int g = 0; g < 4; ++g)
      #pragma unroll
      for (int r = 0; r < 8; ++r) acc[g][r] = bvs[g];

    // ---- input projection: K = 256, 8 k-steps, Wx1 from LDS
    const _Float16* rowx = h0_seq + ((size_t)(m0 + ln) * Tt + td) * 256;
    #pragma unroll
    for (int kk = 0; kk < 8; ++kk) {
      const int k0 = kk * 32;
      v16h a = load_a16(rowx, k0, lh);
      #pragma unroll
      for (int g = 0; g < 4; ++g) {
        v16h bf = load_b16(sWx, 256, g * 64 + ut * 16, k0, lane);
        acc[g] = wmma_f16(a, bf, acc[g]);
      }
    }
    // ---- recurrent: K = 64, 2 k-steps, Wh1 from global (L0/L2-hot)
    #pragma unroll
    for (int kk = 0; kk < 2; ++kk) {
      const int k0 = kk * 32;
      v16h a = load_a16(sH + (size_t)(m0 + ln) * 64, k0, lh);
      #pragma unroll
      for (int g = 0; g < 4; ++g) {
        v16h bf = load_b16(whT, 64, g * 64 + ut * 16, k0, lane);
        acc[g] = wmma_f16(a, bf, acc[g]);
      }
    }
    __syncthreads();

    const int mb   = m0 + 8 * lh;
    const int ncol = ut * 16 + ln;
    #pragma unroll
    for (int r = 0; r < 8; ++r) {
      const float ig = sigm(acc[0][r]);
      const float fg = sigm(acc[1][r]);
      const float gg = fast_tanh(acc[2][r]);
      const float og = sigm(acc[3][r]);
      const float cc = fg * c[r] + ig * gg;
      c[r] = cc;
      const float hv = og * fast_tanh(cc);
      sH[(size_t)(mb + r) * 64 + ncol] = (_Float16)hv;
      if (t == Tt - 1)
        h1_out[(size_t)dir * 4096 + (size_t)(mb + r) * 64 + ncol] = hv;
    }
    __syncthreads();
  }
}

// ---------------------------------------------------------------------------
// MLP head: [64x128] -> relu 128 -> relu 64 -> sigmoid 1. Tiny; one block.
// ---------------------------------------------------------------------------
__global__ __launch_bounds__(128) void head_kernel(
    const float* __restrict__ h1_out,  // [2][64][64]
    const float* __restrict__ d0W, const float* __restrict__ d0b,
    const float* __restrict__ d1W, const float* __restrict__ d1b,
    const float* __restrict__ oW,  const float* __restrict__ ob,
    float* __restrict__ out)           // [64]
{
  __shared__ float sh[64 * 128];
  __shared__ float s0[64 * 128];
  __shared__ float s1[64 * 64];
  const int tid = threadIdx.x;

  for (int i = tid; i < 64 * 128; i += 128) {
    const int b = i >> 7, j = i & 127;
    sh[i] = (j < 64) ? h1_out[b * 64 + j] : h1_out[4096 + b * 64 + (j - 64)];
  }
  __syncthreads();
  for (int i = tid; i < 64 * 128; i += 128) {
    const int b = i >> 7, j = i & 127;
    float s = d0b[j];
    for (int k = 0; k < 128; ++k) s += sh[b * 128 + k] * d0W[k * 128 + j];
    s0[i] = fmaxf(s, 0.0f);
  }
  __syncthreads();
  for (int i = tid; i < 64 * 64; i += 128) {
    const int b = i >> 6, j = i & 63;
    float s = d1b[j];
    for (int k = 0; k < 128; ++k) s += s0[b * 128 + k] * d1W[k * 64 + j];
    s1[i] = fmaxf(s, 0.0f);
  }
  __syncthreads();
  if (tid < 64) {
    float s = ob[0];
    for (int k = 0; k < 64; ++k) s += s1[tid * 64 + k] * oW[k];
    out[tid] = sigm(s);
  }
}

// ---------------------------------------------------------------------------
// Launcher
// ---------------------------------------------------------------------------
extern "C" void kernel_launch(void* const* d_in, const int* in_sizes, int n_in,
                              void* d_out, int out_size, void* d_ws,
                              size_t ws_size, hipStream_t stream) {
  const float* x      = (const float*)d_in[0];
  const float* l0f_Wx = (const float*)d_in[1];
  const float* l0f_Wh = (const float*)d_in[2];
  const float* l0f_b  = (const float*)d_in[3];
  const float* l0b_Wx = (const float*)d_in[4];
  const float* l0b_Wh = (const float*)d_in[5];
  const float* l0b_b  = (const float*)d_in[6];
  const float* l1f_Wx = (const float*)d_in[7];
  const float* l1f_Wh = (const float*)d_in[8];
  const float* l1f_b  = (const float*)d_in[9];
  const float* l1b_Wx = (const float*)d_in[10];
  const float* l1b_Wh = (const float*)d_in[11];
  const float* l1b_b  = (const float*)d_in[12];
  const float* d0_W   = (const float*)d_in[13];
  const float* d0_b   = (const float*)d_in[14];
  const float* d1_W   = (const float*)d_in[15];
  const float* d1_b   = (const float*)d_in[16];
  const float* out_W  = (const float*)d_in[17];
  const float* out_b  = (const float*)d_in[18];

  char* ws = (char*)d_ws;
  _Float16* xT   = (_Float16*)(ws + OFF_XT);
  _Float16* h0   = (_Float16*)(ws + OFF_H0);
  _Float16* wh0T = (_Float16*)(ws + OFF_WH0);
  _Float16* wx0T = (_Float16*)(ws + OFF_WX0);
  _Float16* wx1T = (_Float16*)(ws + OFF_WX1);
  _Float16* wh1T = (_Float16*)(ws + OFF_WH1);
  float*    h1   = (float*)   (ws + OFF_H1);

  // 1) input repack f32 -> f16 [B][T][32]
  prep_x_kernel<<<(Bb * Tt + 255) / 256, 256, 0, stream>>>(x, xT);

  // 2) weight transposes -> N-major f16
  auto tgrid = [](int n) { return (n + 255) / 256; };
  transpose_cast_kernel<<<tgrid(512 * 128), 256, 0, stream>>>(l0f_Wh, wh0T,                 128, 512, 128);
  transpose_cast_kernel<<<tgrid(512 * 128), 256, 0, stream>>>(l0b_Wh, wh0T + 512 * 128,    128, 512, 128);
  transpose_cast_kernel<<<tgrid(512 * 32),  256, 0, stream>>>(l0f_Wx, wx0T,                  22, 512, 32);
  transpose_cast_kernel<<<tgrid(512 * 32),  256, 0, stream>>>(l0b_Wx, wx0T + 512 * 32,       22, 512, 32);
  transpose_cast_kernel<<<tgrid(256 * 256), 256, 0, stream>>>(l1f_Wx, wx1T,                 256, 256, 256);
  transpose_cast_kernel<<<tgrid(256 * 256), 256, 0, stream>>>(l1b_Wx, wx1T + 256 * 256,     256, 256, 256);
  transpose_cast_kernel<<<tgrid(256 * 64),  256, 0, stream>>>(l1f_Wh, wh1T,                  64, 256, 64);
  transpose_cast_kernel<<<tgrid(256 * 64),  256, 0, stream>>>(l1b_Wh, wh1T + 256 * 64,       64, 256, 64);

  // 3) layer-0 bi-LSTM (fwd/bwd concurrently on separate WGPs)
  lstm0_kernel<<<2, 512, 0, stream>>>(xT, wh0T, wx0T, l0f_b, l0b_b, h0);

  // 4) layer-1 bi-LSTM, last states only
  lstm1_kernel<<<2, 512, 0, stream>>>(h0, wx1T, wh1T, l1f_b, l1b_b, h1);

  // 5) dense head -> d_out [64] f32
  head_kernel<<<1, 128, 0, stream>>>(h1, d0_W, d0_b, d1_W, d1_b, out_W, out_b,
                                     (float*)d_out);
}
